// BipartPool_7344394076202
// MI455X (gfx1250) — matrix operand order repaired and checked by reading, not compiled, not run on hardware
//
#include <hip/hip_runtime.h>
#include <hip/hip_bf16.h>

// ---------------------------------------------------------------------------
// Problem constants: B=64 graphs, N=128 nodes/graph, RATIO=32 queries/graph,
// H=8 heads, HD=64, E=512, S=8192, L=2048
// ---------------------------------------------------------------------------
#define EDIM 512
#define SROWS 8192
#define NGRAPH 64
#define NHEAD 8
#define HDIM 64
#define NNODE 128
#define NQ 32

typedef __attribute__((ext_vector_type(16))) __bf16    v16bf;
typedef __attribute__((ext_vector_type(8)))  float     v8f;
typedef __attribute__((ext_vector_type(8)))  unsigned  v8u;
typedef __attribute__((ext_vector_type(4)))  unsigned  v4u;

// round-to-nearest-even f32 -> bf16, pack two into one dword
__device__ __forceinline__ unsigned pk_bf16(float a, float b) {
    unsigned ua = __builtin_bit_cast(unsigned, a);
    unsigned ub = __builtin_bit_cast(unsigned, b);
    ua += 0x7FFFu + ((ua >> 16) & 1u);
    ub += 0x7FFFu + ((ub >> 16) & 1u);
    return (ua >> 16) | (ub & 0xFFFF0000u);
}
__device__ __forceinline__ unsigned short bf16_1(float a) {
    unsigned ua = __builtin_bit_cast(unsigned, a);
    ua += 0x7FFFu + ((ua >> 16) & 1u);
    return (unsigned short)(ua >> 16);
}
__device__ __forceinline__ v8f wmma_bf16(v8u au, v8u bu, v8f c) {
    v16bf a = __builtin_bit_cast(v16bf, au);
    v16bf b = __builtin_bit_cast(v16bf, bu);
    return __builtin_amdgcn_wmma_f32_16x16x32_bf16(
        /*neg_a=*/false, a, /*neg_b=*/false, b,
        /*c_mod=*/(short)0, c, /*reuse_a=*/false, /*reuse_b=*/false);
}

// ISA 16-bit fragment layouts (wave32), lane = hi*16+lo:
//  A (16x32): lane row = lo; VGPRs 0..3 hold K = hi*8+{0,2,4,6}(+1),
//             VGPRs 4..7 hold K = 16+hi*8+{0,2,4,6}(+1)
//             -> two contiguous 16B runs at k = hi*8 and k = 16+hi*8
//  B (32x16): lane col = lo; same K order as A
//             -> two contiguous 16B runs at k = hi*8 and k = 16+hi*8
__device__ __forceinline__ v8u load_afrag(const unsigned short* base, int hi) {
    v8u u;
    u.lo = *(const v4u*)(base + hi * 8);
    u.hi = *(const v4u*)(base + 16 + hi * 8);
    return u;
}
__device__ __forceinline__ v8u load_bfrag(const unsigned short* base, int hi) {
    v8u u;
    u.lo = *(const v4u*)(base + hi * 8);
    u.hi = *(const v4u*)(base + 16 + hi * 8);
    return u;
}

// ---------------------------------------------------------------------------
// One-time f32 -> bf16 conversion (memory-bound, ~1us total at 23.3 TB/s).
// ---------------------------------------------------------------------------
__global__ void cvt_bf16_kernel(const float* __restrict__ in,
                                unsigned short* __restrict__ out, int n8) {
    int t = blockIdx.x * blockDim.x + threadIdx.x;
    if (t >= n8) return;
    const float4* p = (const float4*)in + (size_t)t * 2;
    float4 f0 = p[0], f1 = p[1];
    uint4 r;
    r.x = pk_bf16(f0.x, f0.y);
    r.y = pk_bf16(f0.z, f0.w);
    r.z = pk_bf16(f1.x, f1.y);
    r.w = pk_bf16(f1.z, f1.w);
    ((uint4*)out)[t] = r;
}

// ---------------------------------------------------------------------------
// out[m*sm + n*sn](bf16) = A[M x 512](bf16) @ W[512 x 512](bf16)^T + bias
// Each wave: one 16x64 strip (1 A-frag reused by 4 B-frags -> 4 WMMA).
// Software-pipelined one k-step ahead with explicit ping-pong register sets
// (unroll-by-2) so no inter-iteration register copies are needed.
// ---------------------------------------------------------------------------
__global__ void proj_kernel(const unsigned short* __restrict__ A,
                            const unsigned short* __restrict__ W,
                            const float* __restrict__ bias,
                            unsigned short* __restrict__ out,
                            int sm, int sn) {
    const int lane = threadIdx.x & 31;
    const int wave = threadIdx.x >> 5;
    const int grp  = blockIdx.x * 8 + wave;
    const int mt   = grp >> 3;
    const int nb   = (grp & 7) * 4;              // first of 4 n-tiles
    const int hi = lane >> 4, lo = lane & 15;

    const unsigned short* arow = A + (size_t)(mt * 16 + lo) * EDIM;
    const unsigned short* wr0  = W + (size_t)((nb + 0) * 16 + lo) * EDIM;
    const unsigned short* wr1  = W + (size_t)((nb + 1) * 16 + lo) * EDIM;
    const unsigned short* wr2  = W + (size_t)((nb + 2) * 16 + lo) * EDIM;
    const unsigned short* wr3  = W + (size_t)((nb + 3) * 16 + lo) * EDIM;

    // ping-pong fragment sets
    v8u auA, b0A, b1A, b2A, b3A;   // set A
    v8u auB, b0B, b1B, b2B, b3B;   // set B

    // prologue: k-step 0 into set A
    auA = load_afrag(arow, hi);
    b0A = load_bfrag(wr0, hi);
    b1A = load_bfrag(wr1, hi);
    b2A = load_bfrag(wr2, hi);
    b3A = load_bfrag(wr3, hi);

    v8f acc0 = {}, acc1 = {}, acc2 = {}, acc3 = {};
    // 7 double-steps: covers k-steps 1..14 loads, 0..13 WMMAs
#pragma unroll 1
    for (int k0 = 32; k0 < EDIM - 64; k0 += 64) {
        // load step s into set B while consuming set A
        auB = load_afrag(arow + k0, hi);
        b0B = load_bfrag(wr0 + k0, hi);
        b1B = load_bfrag(wr1 + k0, hi);
        b2B = load_bfrag(wr2 + k0, hi);
        b3B = load_bfrag(wr3 + k0, hi);
        __builtin_prefetch(arow + k0 + 96, 0, 1);   // global_prefetch_b8
        acc0 = wmma_bf16(auA, b0A, acc0);
        acc1 = wmma_bf16(auA, b1A, acc1);
        acc2 = wmma_bf16(auA, b2A, acc2);
        acc3 = wmma_bf16(auA, b3A, acc3);
        // load step s+1 into set A while consuming set B
        auA = load_afrag(arow + k0 + 32, hi);
        b0A = load_bfrag(wr0 + k0 + 32, hi);
        b1A = load_bfrag(wr1 + k0 + 32, hi);
        b2A = load_bfrag(wr2 + k0 + 32, hi);
        b3A = load_bfrag(wr3 + k0 + 32, hi);
        acc0 = wmma_bf16(auB, b0B, acc0);
        acc1 = wmma_bf16(auB, b1B, acc1);
        acc2 = wmma_bf16(auB, b2B, acc2);
        acc3 = wmma_bf16(auB, b3B, acc3);
    }
    // epilogue: load final k-step (15) into set B, consume set A (14), then B
    {
        const int kf = EDIM - 32;
        auB = load_afrag(arow + kf, hi);
        b0B = load_bfrag(wr0 + kf, hi);
        b1B = load_bfrag(wr1 + kf, hi);
        b2B = load_bfrag(wr2 + kf, hi);
        b3B = load_bfrag(wr3 + kf, hi);
        acc0 = wmma_bf16(auA, b0A, acc0);
        acc1 = wmma_bf16(auA, b1A, acc1);
        acc2 = wmma_bf16(auA, b2A, acc2);
        acc3 = wmma_bf16(auA, b3A, acc3);
        acc0 = wmma_bf16(auB, b0B, acc0);
        acc1 = wmma_bf16(auB, b1B, acc1);
        acc2 = wmma_bf16(auB, b2B, acc2);
        acc3 = wmma_bf16(auB, b3B, acc3);
    }

    const int mbase = mt * 16 + hi * 8;
#pragma unroll
    for (int c = 0; c < 4; ++c) {
        v8f acc = c == 0 ? acc0 : c == 1 ? acc1 : c == 2 ? acc2 : acc3;
        const int n = (nb + c) * 16 + lo;
        const float bv = bias[n];
#pragma unroll
        for (int r = 0; r < 8; ++r)
            out[(size_t)(mbase + r) * sm + (size_t)n * sn] = bf16_1(acc[r] + bv);
    }
}

// ---------------------------------------------------------------------------
// One block per (graph b, head h).  Block-diagonal attention:
//   scores = Q(32x64) @ K_b^T(64x128) / 8  -> LDS f32
//   softmax rows (all 128 keys of graph b allowed by the mask structure)
//   out    = attn(32x128 bf16, LDS) @ V_b(128x64, from transposed Vt) -> AO
// ---------------------------------------------------------------------------
__global__ void attn_kernel(const unsigned short* __restrict__ Qb,
                            const unsigned short* __restrict__ Kb,
                            const unsigned short* __restrict__ Vt,  // [col][node]
                            unsigned short* __restrict__ AO) {
    __shared__ __align__(16) float sc[NQ * NNODE];           // 16 KB
    __shared__ __align__(16) unsigned short at[NQ * NNODE];  // 8 KB

    const int b = blockIdx.x >> 3;
    const int h = blockIdx.x & 7;
    const int lane = threadIdx.x & 31;
    const int wave = threadIdx.x >> 5;
    const int hi = lane >> 4, lo = lane & 15;

    // ---- scores: 2 x 8 tiles of 16x16, K-dim = 64 (2 wmma steps) ----
    for (int t = wave; t < 16; t += 8) {
        const int mt = t >> 3, nt = t & 7;
        const unsigned short* qrow = Qb + (size_t)(mt * 16 + lo) * EDIM + h * HDIM;
        const unsigned short* krow =
            Kb + (size_t)(b * NNODE + nt * 16 + lo) * EDIM + h * HDIM;
        v8f acc = {};
#pragma unroll
        for (int ks = 0; ks < HDIM; ks += 32)
            acc = wmma_bf16(load_afrag(qrow + ks, hi), load_bfrag(krow + ks, hi), acc);
#pragma unroll
        for (int r = 0; r < 8; ++r)
            sc[(mt * 16 + hi * 8 + r) * NNODE + nt * 16 + lo] = acc[r] * 0.125f;
    }
    __syncthreads();

    // ---- softmax: one thread per query row ----
    if (threadIdx.x < NQ) {
        float* row = sc + threadIdx.x * NNODE;
        float mx = row[0];
        for (int j = 1; j < NNODE; ++j) mx = fmaxf(mx, row[j]);
        float s = 0.f;
        for (int j = 0; j < NNODE; ++j) {
            float e = __expf(row[j] - mx);
            row[j] = e;
            s += e;
        }
        float inv = 1.f / s;
        unsigned short* arow = at + threadIdx.x * NNODE;
        for (int j = 0; j < NNODE; ++j) arow[j] = bf16_1(row[j] * inv);
    }
    __syncthreads();

    // ---- out = attn(32x128) @ V(128x64): 2x4 tiles, one per wave ----
    {
        const int mt = wave >> 2, nt = wave & 3;
        const unsigned short* arow = at + (mt * 16 + lo) * NNODE;
        // B element (k=node, n=feat) = Vt[(h*64 + n) * SROWS + b*128 + k]
        const unsigned short* vrow =
            Vt + (size_t)(h * HDIM + nt * 16 + lo) * SROWS + b * NNODE;
        v8f acc = {};
#pragma unroll
        for (int ks = 0; ks < NNODE; ks += 32)
            acc = wmma_bf16(load_afrag(arow + ks, hi), load_bfrag(vrow + ks, hi), acc);
#pragma unroll
        for (int r = 0; r < 8; ++r) {
            int q = b * NQ + mt * 16 + hi * 8 + r;
            int n = h * HDIM + nt * 16 + lo;
            AO[(size_t)q * EDIM + n] = bf16_1(acc[r]);
        }
    }
}

// ---------------------------------------------------------------------------
// d_out[2048 x 512](f32) = AO(bf16) @ Wo(bf16)^T + bias; same ping-pong
// pipelined 16x64-strip structure as proj_kernel.
// ---------------------------------------------------------------------------
__global__ void outproj_kernel(const unsigned short* __restrict__ AO,
                               const unsigned short* __restrict__ Wo,
                               const float* __restrict__ bias,
                               float* __restrict__ out) {
    const int lane = threadIdx.x & 31;
    const int wave = threadIdx.x >> 5;
    const int grp  = blockIdx.x * 8 + wave;
    const int mt   = grp >> 3;
    const int nb   = (grp & 7) * 4;
    const int hi = lane >> 4, lo = lane & 15;

    const unsigned short* arow = AO + (size_t)(mt * 16 + lo) * EDIM;
    const unsigned short* wr0  = Wo + (size_t)((nb + 0) * 16 + lo) * EDIM;
    const unsigned short* wr1  = Wo + (size_t)((nb + 1) * 16 + lo) * EDIM;
    const unsigned short* wr2  = Wo + (size_t)((nb + 2) * 16 + lo) * EDIM;
    const unsigned short* wr3  = Wo + (size_t)((nb + 3) * 16 + lo) * EDIM;

    v8u auA, b0A, b1A, b2A, b3A;
    v8u auB, b0B, b1B, b2B, b3B;

    auA = load_afrag(arow, hi);
    b0A = load_bfrag(wr0, hi);
    b1A = load_bfrag(wr1, hi);
    b2A = load_bfrag(wr2, hi);
    b3A = load_bfrag(wr3, hi);

    v8f acc0 = {}, acc1 = {}, acc2 = {}, acc3 = {};
#pragma unroll 1
    for (int k0 = 32; k0 < EDIM - 64; k0 += 64) {
        auB = load_afrag(arow + k0, hi);
        b0B = load_bfrag(wr0 + k0, hi);
        b1B = load_bfrag(wr1 + k0, hi);
        b2B = load_bfrag(wr2 + k0, hi);
        b3B = load_bfrag(wr3 + k0, hi);
        acc0 = wmma_bf16(auA, b0A, acc0);
        acc1 = wmma_bf16(auA, b1A, acc1);
        acc2 = wmma_bf16(auA, b2A, acc2);
        acc3 = wmma_bf16(auA, b3A, acc3);
        auA = load_afrag(arow + k0 + 32, hi);
        b0A = load_bfrag(wr0 + k0 + 32, hi);
        b1A = load_bfrag(wr1 + k0 + 32, hi);
        b2A = load_bfrag(wr2 + k0 + 32, hi);
        b3A = load_bfrag(wr3 + k0 + 32, hi);
        acc0 = wmma_bf16(auB, b0B, acc0);
        acc1 = wmma_bf16(auB, b1B, acc1);
        acc2 = wmma_bf16(auB, b2B, acc2);
        acc3 = wmma_bf16(auB, b3B, acc3);
    }
    {
        const int kf = EDIM - 32;
        auB = load_afrag(arow + kf, hi);
        b0B = load_bfrag(wr0 + kf, hi);
        b1B = load_bfrag(wr1 + kf, hi);
        b2B = load_bfrag(wr2 + kf, hi);
        b3B = load_bfrag(wr3 + kf, hi);
        acc0 = wmma_bf16(auA, b0A, acc0);
        acc1 = wmma_bf16(auA, b1A, acc1);
        acc2 = wmma_bf16(auA, b2A, acc2);
        acc3 = wmma_bf16(auA, b3A, acc3);
        acc0 = wmma_bf16(auB, b0B, acc0);
        acc1 = wmma_bf16(auB, b1B, acc1);
        acc2 = wmma_bf16(auB, b2B, acc2);
        acc3 = wmma_bf16(auB, b3B, acc3);
    }

    const int mbase = mt * 16 + hi * 8;
#pragma unroll
    for (int c = 0; c < 4; ++c) {
        v8f acc = c == 0 ? acc0 : c == 1 ? acc1 : c == 2 ? acc2 : acc3;
        const int n = (nb + c) * 16 + lo;
        const float bv = bias[n];
#pragma unroll
        for (int r = 0; r < 8; ++r)
            out[(size_t)(mbase + r) * EDIM + n] = acc[r] + bv;
    }
}

// ---------------------------------------------------------------------------
// Host side.
// Inputs: 0=x(64,128,512) 1=batch(8192,i32; unused: mask is block-diagonal)
//         2=aggrs(32,512) 3=in_proj_w(1536,512) 4=in_proj_b(1536)
//         5=out_proj_w(512,512) 6=out_proj_b(512)
// ---------------------------------------------------------------------------
extern "C" void kernel_launch(void* const* d_in, const int* in_sizes, int n_in,
                              void* d_out, int out_size, void* d_ws, size_t ws_size,
                              hipStream_t stream) {
    const float* x     = (const float*)d_in[0];
    const float* aggrs = (const float*)d_in[2];
    const float* ipw   = (const float*)d_in[3];
    const float* ipb   = (const float*)d_in[4];
    const float* opw   = (const float*)d_in[5];
    const float* opb   = (const float*)d_in[6];
    float* out         = (float*)d_out;

    // workspace layout (ushort units); total ~29.4 MB, all L2-resident
    unsigned short* xb   = (unsigned short*)d_ws;                 // 8192*512
    unsigned short* ipwb = xb   + (size_t)SROWS * EDIM;           // 1536*512
    unsigned short* opwb = ipwb + (size_t)3 * EDIM * EDIM;        //  512*512
    unsigned short* aggb = opwb + (size_t)EDIM * EDIM;            //   32*512
    unsigned short* Qb   = aggb + (size_t)NQ * EDIM;              //   32*512
    unsigned short* Kb   = Qb   + (size_t)NQ * EDIM;              // 8192*512
    unsigned short* Vt   = Kb   + (size_t)SROWS * EDIM;           //  512*8192 (transposed)
    unsigned short* AO   = Vt   + (size_t)EDIM * SROWS;           // 2048*512

    const unsigned short* Wqb = ipwb;
    const unsigned short* Wkb = ipwb + (size_t)EDIM * EDIM;
    const unsigned short* Wvb = ipwb + (size_t)2 * EDIM * EDIM;
    const float* bq = ipb;
    const float* bk = ipb + EDIM;
    const float* bv = ipb + 2 * EDIM;

    // one-time f32 -> bf16 conversions
    cvt_bf16_kernel<<<(SROWS * EDIM / 8 + 255) / 256, 256, 0, stream>>>(x, xb, SROWS * EDIM / 8);
    cvt_bf16_kernel<<<(3 * EDIM * EDIM / 8 + 255) / 256, 256, 0, stream>>>(ipw, ipwb, 3 * EDIM * EDIM / 8);
    cvt_bf16_kernel<<<(EDIM * EDIM / 8 + 255) / 256, 256, 0, stream>>>(opw, opwb, EDIM * EDIM / 8);
    cvt_bf16_kernel<<<(NQ * EDIM / 8 + 255) / 256, 256, 0, stream>>>(aggrs, aggb, NQ * EDIM / 8);

    // projections (16x64 strip per wave, 8 waves/block)
    proj_kernel<<<2, 256, 0, stream>>>(aggb, Wqb, bq, Qb, EDIM, 1);     //   32x512
    proj_kernel<<<512, 256, 0, stream>>>(xb, Wkb, bk, Kb, EDIM, 1);     // 8192x512
    proj_kernel<<<512, 256, 0, stream>>>(xb, Wvb, bv, Vt, 1, SROWS);    // 8192x512, transposed out

    // attention: one block per (graph, head)
    attn_kernel<<<NGRAPH * NHEAD, 256, 0, stream>>>(Qb, Kb, Vt, AO);

    // output projection: 2048x512
    outproj_kernel<<<128, 256, 0, stream>>>(AO, opwb, opb, out);
}
